// VisionGNN_73332271612088
// MI455X (gfx1250) — matrix operand order repaired and checked by reading, not compiled, not run on hardware
//
#include <hip/hip_runtime.h>
#include <hip/hip_bf16.h>
#include <stdint.h>

// ---------- CDNA5 WMMA types (wave32, 16x16x32 bf16 -> f32) ----------
typedef __attribute__((ext_vector_type(16))) __bf16 v16bf;
typedef __attribute__((ext_vector_type(8)))  float  v8f;

union Frag { int4 i4[2]; v16bf v; };
union Acc  { v8f v; float f[8]; };

__device__ __forceinline__ unsigned short f2bf(float x) {
  unsigned int u = __builtin_bit_cast(unsigned int, x);
  u += 0x7FFFu + ((u >> 16) & 1u);          // round-to-nearest-even
  return (unsigned short)(u >> 16);
}
__device__ __forceinline__ float bf2f(unsigned short h) {
  return __builtin_bit_cast(float, ((unsigned int)h) << 16);
}

#define AP 72   // LDS pitch in halves: 64 K + 8 pad (16B-aligned rows, no bank alias)

// =====================================================================
// One-time weight prep: W[k][n] f32  ->  Wt[n][k] bf16 (pre-transposed)
// =====================================================================
__global__ __launch_bounds__(256)
void wprep(const float* __restrict__ W, unsigned short* __restrict__ Wt, int K) {
  const int idx = blockIdx.x * 256 + threadIdx.x;   // over K*128
  const int k = idx >> 7, n = idx & 127;
  Wt[(long)n * K + k] = f2bf(W[(long)k * 128 + n]);
}

// =====================================================================
// Embedding GEMM with fused patch gather:
//   H[node, :] = bf16( patches[node] @ emb_W + emb_b )
// M = B*1024, K = 1280 (20 channels x 64), N = 128.
// Block = 256 thr (8 waves); tile 256(M) x 128(N); wave owns 32 rows
// (two 16-row WMMA strips -> each B fragment feeds 2 WMMAs).
// B fragments are register double-buffered to overlap LDS latency.
// K-chunk = 64 = exactly one input channel.
// =====================================================================
__global__ __launch_bounds__(256)
void emb_gemm(const float* __restrict__ x, const unsigned short* __restrict__ Wt,
              const float* __restrict__ bias, unsigned short* __restrict__ H) {
  __shared__ unsigned short As[256 * AP];   // A tile, row-major [m][k]
  __shared__ unsigned short Bs[128 * AP];   // B tile, transposed [n][k]

  const int tid  = threadIdx.x;
  const int lane = tid & 31, wave = tid >> 5;
  const int hi   = lane >> 4, l15 = lane & 15;
  const int blockRow = blockIdx.x * 256;

  // gather geometry: one thread owns one node row of the A tile
  const int node = blockRow + tid;
  const int b  = node >> 10, n = node & 1023;
  const int ny = n >> 5,     nx = n & 31;
  const long xbase = (long)b * 20 * 65536 + (long)ny * 2048 + nx * 8;

  Acc acc0[8], acc1[8];
  #pragma unroll
  for (int t = 0; t < 8; ++t)
    #pragma unroll
    for (int r = 0; r < 8; ++r) { acc0[t].f[r] = 0.f; acc1[t].f[r] = 0.f; }

  const int m0 = wave * 32, m1 = wave * 32 + 16;

  for (int c = 0; c < 20; ++c) {
    const float* xc = x + xbase + (long)c * 65536;
    if (c + 1 < 20) __builtin_prefetch(xc + 65536, 0, 0);   // next channel
    // ---- A tile: 256 nodes x 64 K (channel c), gathered + f32->bf16 ----
    #pragma unroll
    for (int py = 0; py < 8; ++py) {
      const float4 fa = *(const float4*)(xc + py * 256);
      const float4 fb = *(const float4*)(xc + py * 256 + 4);
      const unsigned int p0 = (unsigned)f2bf(fa.x) | ((unsigned)f2bf(fa.y) << 16);
      const unsigned int p1 = (unsigned)f2bf(fa.z) | ((unsigned)f2bf(fa.w) << 16);
      const unsigned int p2 = (unsigned)f2bf(fb.x) | ((unsigned)f2bf(fb.y) << 16);
      const unsigned int p3 = (unsigned)f2bf(fb.z) | ((unsigned)f2bf(fb.w) << 16);
      *(int4*)&As[tid * AP + py * 8] = make_int4((int)p0, (int)p1, (int)p2, (int)p3);
    }
    // ---- B tile: pre-transposed bf16 weights, straight b128 copies ----
    #pragma unroll
    for (int it = 0; it < 4; ++it) {
      const int idx = tid + it * 256;
      const int nc = idx >> 3, q = idx & 7;
      *(int4*)&Bs[nc * AP + q * 8] =
          *(const int4*)&Wt[(long)nc * 1280 + c * 64 + q * 8];
    }
    __syncthreads();

    // ---- compute: two K=32 WMMA steps, 8 N-tiles x 2 M-strips per wave ----
    #pragma unroll
    for (int ks = 0; ks < 2; ++ks) {
      Frag a0, a1;
      const int kab = ks * 32 + hi * 8;      // A: lanes16-31 hold K+8 / K+24 halves
      a0.i4[0] = *(const int4*)&As[(m0 + l15) * AP + kab];
      a0.i4[1] = *(const int4*)&As[(m0 + l15) * AP + kab + 16];
      a1.i4[0] = *(const int4*)&As[(m1 + l15) * AP + kab];
      a1.i4[1] = *(const int4*)&As[(m1 + l15) * AP + kab + 16];
      const int kbb = ks * 32 + hi * 8;      // B: lanes16-31 hold K+8 / K+24 halves
      Frag bfr[2];
      bfr[0].i4[0] = *(const int4*)&Bs[l15 * AP + kbb];
      bfr[0].i4[1] = *(const int4*)&Bs[l15 * AP + kbb + 16];
      #pragma unroll
      for (int t = 0; t < 8; ++t) {
        const int cur = t & 1, nxt = cur ^ 1;
        if (t < 7) {                         // prefetch next B fragment
          bfr[nxt].i4[0] = *(const int4*)&Bs[((t + 1) * 16 + l15) * AP + kbb];
          bfr[nxt].i4[1] = *(const int4*)&Bs[((t + 1) * 16 + l15) * AP + kbb + 16];
        }
        acc0[t].v = __builtin_amdgcn_wmma_f32_16x16x32_bf16(
            false, a0.v, false, bfr[cur].v, (short)0, acc0[t].v, false, false);
        acc1[t].v = __builtin_amdgcn_wmma_f32_16x16x32_bf16(
            false, a1.v, false, bfr[cur].v, (short)0, acc1[t].v, false, false);
      }
    }
    __syncthreads();
  }

  // epilogue: +bias, f32 -> bf16
  #pragma unroll
  for (int t = 0; t < 8; ++t) {
    const int col = t * 16 + l15;
    const float bv = bias[col];
    #pragma unroll
    for (int r = 0; r < 8; ++r) {
      const int mm = r + hi * 8;             // C/D layout: M = r + 8*hi
      H[(long)(blockRow + m0 + mm) * 128 + col] = f2bf(acc0[t].f[r] + bv);
      H[(long)(blockRow + m1 + mm) * 128 + col] = f2bf(acc1[t].f[r] + bv);
    }
  }
}

// =====================================================================
// GCN GEMM, in-place on H:  acc = Hio[tile] @ Wt.  Tile 256x128 as above.
//  rows >= 1024 (not graph-mixed): Hio <- bf16(relu(acc + bias))  [fused]
//  rows <  1024 (batch-0 block) : Y0 <- acc (raw f32, mixed+activated later)
// Each block touches only its own 256 rows of Hio -> in-place is race-free.
// =====================================================================
__global__ __launch_bounds__(256)
void gcn_gemm(unsigned short* Hio, const unsigned short* __restrict__ Wt,
              const float* __restrict__ bias, float* __restrict__ Y0) {
  __shared__ unsigned short As[256 * AP];
  __shared__ unsigned short Bs[128 * AP];

  const int tid  = threadIdx.x;
  const int lane = tid & 31, wave = tid >> 5;
  const int hi   = lane >> 4, l15 = lane & 15;
  const int blockRow = blockIdx.x * 256;

  Acc acc0[8], acc1[8];
  #pragma unroll
  for (int t = 0; t < 8; ++t)
    #pragma unroll
    for (int r = 0; r < 8; ++r) { acc0[t].f[r] = 0.f; acc1[t].f[r] = 0.f; }

  const int m0 = wave * 32, m1 = wave * 32 + 16;

  for (int kc = 0; kc < 2; ++kc) {
    // A: 256 rows x 64 halves, b128 copies
    #pragma unroll
    for (int it = 0; it < 8; ++it) {
      const int idx = tid + it * 256;
      const int row = idx >> 3, q = idx & 7;
      *(int4*)&As[row * AP + q * 8] =
          *(const int4*)&Hio[(long)(blockRow + row) * 128 + kc * 64 + q * 8];
    }
    // B: pre-transposed bf16 weights, b128 copies
    #pragma unroll
    for (int it = 0; it < 4; ++it) {
      const int idx = tid + it * 256;
      const int nc = idx >> 3, q = idx & 7;
      *(int4*)&Bs[nc * AP + q * 8] =
          *(const int4*)&Wt[(long)nc * 128 + kc * 64 + q * 8];
    }
    __syncthreads();

    #pragma unroll
    for (int ks = 0; ks < 2; ++ks) {
      Frag a0, a1;
      const int kab = ks * 32 + hi * 8;
      a0.i4[0] = *(const int4*)&As[(m0 + l15) * AP + kab];
      a0.i4[1] = *(const int4*)&As[(m0 + l15) * AP + kab + 16];
      a1.i4[0] = *(const int4*)&As[(m1 + l15) * AP + kab];
      a1.i4[1] = *(const int4*)&As[(m1 + l15) * AP + kab + 16];
      const int kbb = ks * 32 + hi * 8;
      Frag bfr[2];
      bfr[0].i4[0] = *(const int4*)&Bs[l15 * AP + kbb];
      bfr[0].i4[1] = *(const int4*)&Bs[l15 * AP + kbb + 16];
      #pragma unroll
      for (int t = 0; t < 8; ++t) {
        const int cur = t & 1, nxt = cur ^ 1;
        if (t < 7) {
          bfr[nxt].i4[0] = *(const int4*)&Bs[((t + 1) * 16 + l15) * AP + kbb];
          bfr[nxt].i4[1] = *(const int4*)&Bs[((t + 1) * 16 + l15) * AP + kbb + 16];
        }
        acc0[t].v = __builtin_amdgcn_wmma_f32_16x16x32_bf16(
            false, a0.v, false, bfr[cur].v, (short)0, acc0[t].v, false, false);
        acc1[t].v = __builtin_amdgcn_wmma_f32_16x16x32_bf16(
            false, a1.v, false, bfr[cur].v, (short)0, acc1[t].v, false, false);
      }
    }
    __syncthreads();
  }

  if (blockRow >= 1024) {       // fused bias + ReLU + bf16, in place
    #pragma unroll
    for (int t = 0; t < 8; ++t) {
      const int col = t * 16 + l15;
      const float bv = bias[col];
      #pragma unroll
      for (int r = 0; r < 8; ++r) {
        const int mm = r + hi * 8;
        Hio[(long)(blockRow + m0 + mm) * 128 + col] = f2bf(fmaxf(acc0[t].f[r] + bv, 0.f));
        Hio[(long)(blockRow + m1 + mm) * 128 + col] = f2bf(fmaxf(acc1[t].f[r] + bv, 0.f));
      }
    }
  } else {                      // batch-0 block: raw f32 for graph mixing
    #pragma unroll
    for (int t = 0; t < 8; ++t) {
      const int col = t * 16 + l15;
      #pragma unroll
      for (int r = 0; r < 8; ++r) {
        const int mm = r + hi * 8;
        Y0[(long)(blockRow + m0 + mm) * 128 + col] = acc0[t].f[r];
        Y0[(long)(blockRow + m1 + mm) * 128 + col] = acc1[t].f[r];
      }
    }
  }
}

// =====================================================================
// Graph mixing on batch-0 block: out[j] = B*dinv[j]*prefix[j] + y[j]/d[j],
// prefix[j] = sum_{i<j} dinv[i]*y[i], d[i] = B*i+1.  One block per column,
// 1024-element exclusive scan (per-thread 4 + Hillis-Steele over 256).
// =====================================================================
__global__ __launch_bounds__(256)
void mix_col(float* __restrict__ Y, int ldc, float Bf) {
  __shared__ float ts[256];
  const int tid = threadIdx.x;
  const int col = blockIdx.x;

  float yv[4], sv[4], dinv[4];
  #pragma unroll
  for (int j = 0; j < 4; ++j) {
    const int i = tid * 4 + j;
    yv[j] = Y[(long)i * ldc + col];
    const float d = Bf * (float)i + 1.0f;
    dinv[j] = rsqrtf(d);
    sv[j] = dinv[j] * yv[j];
  }
  float excl[4], run = 0.f;
  #pragma unroll
  for (int j = 0; j < 4; ++j) { excl[j] = run; run += sv[j]; }
  const float tot = run;
  float val = tot;
  ts[tid] = val; __syncthreads();
  #pragma unroll
  for (int off = 1; off < 256; off <<= 1) {
    const float add = (tid >= off) ? ts[tid - off] : 0.f;
    __syncthreads();
    val += add; ts[tid] = val;
    __syncthreads();
  }
  const float texcl = val - tot;             // exclusive prefix of thread totals
  #pragma unroll
  for (int j = 0; j < 4; ++j) {
    const int i = tid * 4 + j;
    const float d = Bf * (float)i + 1.0f;
    Y[(long)i * ldc + col] = Bf * dinv[j] * (texcl + excl[j]) + yv[j] / d;
  }
}

// bias + ReLU + f32->bf16 for the 1024 mixed batch-0 rows only
__global__ __launch_bounds__(256)
void bias_act_b0(const float* __restrict__ Y0, const float* __restrict__ bias,
                 unsigned short* __restrict__ H) {
  const long i4 = ((long)blockIdx.x * 256 + threadIdx.x) * 4;   // over 1024*128
  const float4 y = *(const float4*)(Y0 + i4);
  const int c0 = (int)(i4 & 127);
  const float v0 = fmaxf(y.x + bias[c0 + 0], 0.f);
  const float v1 = fmaxf(y.y + bias[c0 + 1], 0.f);
  const float v2 = fmaxf(y.z + bias[c0 + 2], 0.f);
  const float v3 = fmaxf(y.w + bias[c0 + 3], 0.f);
  const unsigned int p0 = (unsigned)f2bf(v0) | ((unsigned)f2bf(v1) << 16);
  const unsigned int p1 = (unsigned)f2bf(v2) | ((unsigned)f2bf(v3) << 16);
  *(uint2*)&H[i4] = make_uint2(p0, p1);
}

// Final projection: y[row] = dot(H[row,:128], W2). One wave per row.
__global__ __launch_bounds__(256)
void dot128(const unsigned short* __restrict__ H, const float* __restrict__ W2,
            float* __restrict__ y) {
  const int lane = threadIdx.x & 31, wave = threadIdx.x >> 5;
  const long row = (long)blockIdx.x * 8 + wave;
  const uint2 u = *(const uint2*)&H[row * 128 + lane * 4];
  float s = bf2f((unsigned short)(u.x & 0xFFFFu)) * W2[lane * 4 + 0]
          + bf2f((unsigned short)(u.x >> 16))     * W2[lane * 4 + 1]
          + bf2f((unsigned short)(u.y & 0xFFFFu)) * W2[lane * 4 + 2]
          + bf2f((unsigned short)(u.y >> 16))     * W2[lane * 4 + 3];
  #pragma unroll
  for (int o = 16; o > 0; o >>= 1) s += __shfl_xor(s, o, 32);
  if (lane == 0) y[row] = s;
}

// global mean pool over 1024 nodes per batch, + output bias
__global__ __launch_bounds__(256)
void pool1024(const float* __restrict__ y, const float* __restrict__ b2,
              float* __restrict__ out) {
  __shared__ float red[256];
  const int tid = threadIdx.x;
  const long base = (long)blockIdx.x * 1024;
  float s = 0.f;
  #pragma unroll
  for (int j = 0; j < 4; ++j) s += y[base + tid * 4 + j];
  red[tid] = s; __syncthreads();
  for (int off = 128; off > 0; off >>= 1) {
    if (tid < off) red[tid] += red[tid + off];
    __syncthreads();
  }
  if (tid == 0) out[blockIdx.x] = b2[0] + red[0] * (1.0f / 1024.0f);
}

extern "C" void kernel_launch(void* const* d_in, const int* in_sizes, int n_in,
                              void* d_out, int out_size, void* d_ws, size_t ws_size,
                              hipStream_t stream) {
  const float* x    = (const float*)d_in[0];
  const float* embW = (const float*)d_in[1];
  const float* embB = (const float*)d_in[2];
  const float* W0   = (const float*)d_in[3];
  const float* b0   = (const float*)d_in[4];
  const float* W1   = (const float*)d_in[5];
  const float* b1   = (const float*)d_in[6];
  const float* W2   = (const float*)d_in[7];
  const float* b2   = (const float*)d_in[8];
  float* out = (float*)d_out;

  const int Btch = in_sizes[0] / (20 * 256 * 256);   // = 64
  const int M = Btch * 1024;                          // total nodes

  // ---- workspace carve-up (~17 MB) ----
  char* p = (char*)d_ws;
  unsigned short* Hbf   = (unsigned short*)p;  p += (size_t)M * 128 * 2;   // 16 MB
  unsigned short* embWt = (unsigned short*)p;  p += (size_t)1280 * 128 * 2;
  unsigned short* W0t   = (unsigned short*)p;  p += (size_t)128 * 128 * 2;
  unsigned short* W1t   = (unsigned short*)p;  p += (size_t)128 * 128 * 2;
  float* Y0 = (float*)p;                       p += (size_t)1024 * 128 * 4; // 512 KB
  float* yv = (float*)p;                                                    // M * 4

  // one-time weight transpose+convert to bf16
  wprep<<<(1280 * 128) / 256, 256, 0, stream>>>(embW, embWt, 1280);
  wprep<<<(128  * 128) / 256, 256, 0, stream>>>(W0,   W0t,   128);
  wprep<<<(128  * 128) / 256, 256, 0, stream>>>(W1,   W1t,   128);

  emb_gemm<<<M / 256, 256, 0, stream>>>(x, embWt, embB, Hbf);

  gcn_gemm  <<<M / 256, 256, 0, stream>>>(Hbf, W0t, b0, Y0);
  mix_col   <<<128, 256, 0, stream>>>(Y0, 128, (float)Btch);
  bias_act_b0<<<128, 256, 0, stream>>>(Y0, b0, Hbf);

  gcn_gemm  <<<M / 256, 256, 0, stream>>>(Hbf, W1t, b1, Y0);
  mix_col   <<<128, 256, 0, stream>>>(Y0, 128, (float)Btch);
  bias_act_b0<<<128, 256, 0, stream>>>(Y0, b1, Hbf);

  dot128  <<<M / 8, 256, 0, stream>>>(Hbf, W2, yv);
  mix_col <<<1, 256, 0, stream>>>(yv, 1, (float)Btch);
  pool1024<<<Btch, 256, 0, stream>>>(yv, b2, out);
}
